// VGGBase_8581344657865
// MI455X (gfx1250) — compile-verified, matrix operand order checked by reading, not run on hardware
//
#include <hip/hip_runtime.h>
#include <hip/hip_bf16.h>

// ---------------------------------------------------------------------------
// Types / helpers
// ---------------------------------------------------------------------------
typedef __bf16 bf16;
typedef __attribute__((ext_vector_type(16))) __bf16 v16bf;
typedef __attribute__((ext_vector_type(8)))  float  v8f;

__device__ __forceinline__ bf16 cvt_bf16(float f) {
    unsigned u = __builtin_bit_cast(unsigned, f);
    unsigned r = u + 0x7FFFu + ((u >> 16) & 1u);   // round-to-nearest-even
    unsigned short h = (unsigned short)(r >> 16);
    return __builtin_bit_cast(bf16, h);
}

// Load a 16x32 bf16 WMMA operand fragment from an LDS tile stored
// "row per lane" (row-major, padded stride).  Works for A (rows=M) and for
// B when the tile is stored transposed (rows=N).
//   lanes 0-15  : row = row0+lane,    K = koff + {0..7, 16..23}
//   lanes 16-31 : row = row0+lane-16, K = koff + {8..15, 24..31}
__device__ __forceinline__ v16bf frag_ld(const bf16* s, int stride, int row0,
                                         int koff, int lane) {
    const bf16* p = s + (row0 + (lane & 15)) * stride + koff + ((lane & 16) ? 8 : 0);
    v16bf f;
#pragma unroll
    for (int j = 0; j < 4; ++j) {
        f[2*j]   = p[2*j];
        f[2*j+1] = p[2*j+1];
        f[2*j+8] = p[16 + 2*j];
        f[2*j+9] = p[16 + 2*j+1];
    }
    return f;
}

__device__ __forceinline__ v8f wmma_bf16(v16bf a, v16bf b, v8f c) {
    return __builtin_amdgcn_wmma_f32_16x16x32_bf16(false, a, false, b,
                                                   (short)0, c, false, false);
}

// ---------------------------------------------------------------------------
// Problem constants (match reference setup_inputs)
// ---------------------------------------------------------------------------
#define IMG_H 192
#define IMG_W 192
#define CDIM  256
#define B2    4                    // 2*b stacked feature maps
#define LTOK  (IMG_H * IMG_W)      // 36864
#define MROW  (B2 * LTOK)          // 147456
#define NWIN  16
#define WSZ   12
#define SWIN  (WSZ * WSZ)          // 144 tokens / window
#define SHFT  6

// ---------------------------------------------------------------------------
// fp32 -> bf16 conversion (weights)
// ---------------------------------------------------------------------------
__global__ void k_f2b(const float* __restrict__ x, bf16* __restrict__ y, int n) {
    int i = blockIdx.x * blockDim.x + threadIdx.x;
    if (i < n) y[i] = cvt_bf16(x[i]);
}

// ---------------------------------------------------------------------------
// Pack: (b,c,h,w) fp32 -> feat_rt / feat_tr  [4, L, C] fp32
// ---------------------------------------------------------------------------
__global__ void k_pack(const float* __restrict__ fr, const float* __restrict__ ft,
                       float* __restrict__ A, float* __restrict__ B) {
    size_t i = (size_t)blockIdx.x * blockDim.x + threadIdx.x;
    const size_t total = (size_t)B2 * LTOK * CDIM;
    if (i >= total) return;
    int c  = (int)(i & (CDIM - 1));
    size_t rest = i >> 8;
    int l  = (int)(rest % LTOK);
    int bi = (int)(rest / LTOK);
    int img = bi & 1;
    const float* sA = (bi < 2) ? fr : ft;
    const float* sB = (bi < 2) ? ft : fr;
    size_t sidx = ((size_t)img * CDIM + c) * LTOK + l;
    A[i] = sA[sidx];
    B[i] = sB[sidx];
}

// featB = concat(second half of featA, first half)
__global__ void k_swap(const float* __restrict__ A, float* __restrict__ B) {
    size_t i = (size_t)blockIdx.x * blockDim.x + threadIdx.x;
    const size_t total = (size_t)B2 * LTOK * CDIM;
    const size_t half  = total / 2;
    if (i >= total) return;
    B[i] = A[i < half ? i + half : i - half];
}

// Unpack: featA [4,L,C] -> out_r(2,C,H,W) ++ out_t(2,C,H,W)
__global__ void k_unpack(const float* __restrict__ A, float* __restrict__ out) {
    size_t i = (size_t)blockIdx.x * blockDim.x + threadIdx.x;
    const size_t per = (size_t)2 * CDIM * LTOK;        // elems per output tensor
    const size_t total = 2 * per;
    if (i >= total) return;
    int hs = (int)(i / per);                 // 0 = out_r, 1 = out_t
    size_t rem = i - (size_t)hs * per;
    int b = (int)(rem / ((size_t)CDIM * LTOK));
    size_t r2 = rem % ((size_t)CDIM * LTOK);
    int c = (int)(r2 / LTOK);
    int l = (int)(r2 % LTOK);
    int bi = hs * 2 + b;
    out[i] = A[((size_t)bi * LTOK + l) * CDIM + c];
}

// ---------------------------------------------------------------------------
// Generic bf16 WMMA GEMM:  C[M,N] = op(A[M,K]) @ B[K,N]
//   A: fp32 (Af) or bf16 (Ab);  B: bf16 row-major;  C: fp32 (Cf) or bf16 (Cb)
//   optional exact-GELU epilogue.
// Block tile 128x256, K-step 32, 256 threads (8 waves, 64x64 wave tiles,
// 16 WMMAs/wave/K-step).  Double-buffered LDS with register staging so the
// next tile's global loads are in flight under the WMMA block.
// ---------------------------------------------------------------------------
#define GBM 128
#define GBN 256
#define GBK 32
#define LSTR 40   // padded LDS stride (bank-conflict-free fragment reads)

__global__ __launch_bounds__(256) void
k_gemm(const float* __restrict__ Af, const bf16* __restrict__ Ab,
       const bf16* __restrict__ Bw,
       float* __restrict__ Cf, bf16* __restrict__ Cb,
       int M, int N, int K, int gelu) {
    __shared__ bf16 sA[2][GBM * LSTR];
    __shared__ bf16 sB[2][GBN * LSTR];   // stored transposed: sB[n][k]

    const int tid  = threadIdx.x;
    const int lane = tid & 31;
    const int wid  = tid >> 5;
    const int wm   = wid >> 2;        // 0..1  -> 64-row slice
    const int wn   = wid & 3;         // 0..3  -> 64-col slice
    const size_t gm0 = (size_t)blockIdx.y * GBM;
    const size_t gn0 = (size_t)blockIdx.x * GBN;

    // ---- staging registers ----
    // A: 128x32 tile, 16 elems/thread (4 units of 4).
    const int ar  = (tid + 0 * 256) >> 3;          // base row for unit pattern
    float4 raf[4];                                 // fp32 A path
    unsigned long long rab[4];                     // bf16 A path (4 elems each)
    // B: 32x256 tile, 32 elems/thread: row k, cols n0..n0+31.
    const int bk  = tid >> 3;
    const int bn0 = (tid & 7) << 5;
    uint4 rbv[4];                                  // 4 x 8 bf16

    auto ldg = [&](int k0) {
        if (Af) {
            const float* Ag = Af + gm0 * K + k0;
#pragma unroll
            for (int i = 0; i < 4; ++i) {
                int f = tid + i * 256;
                int r = f >> 3, c4 = (f & 7) << 2;
                raf[i] = *(const float4*)(Ag + (size_t)r * K + c4);
            }
        } else {
            const bf16* Ag = Ab + gm0 * K + k0;
#pragma unroll
            for (int i = 0; i < 4; ++i) {
                int f = tid + i * 256;
                int r = f >> 3, c4 = (f & 7) << 2;
                rab[i] = *(const unsigned long long*)(Ag + (size_t)r * K + c4);
            }
        }
        const bf16* Bg = Bw + ((size_t)k0 + bk) * N + gn0 + bn0;
#pragma unroll
        for (int j = 0; j < 4; ++j) rbv[j] = ((const uint4*)Bg)[j];
    };

    auto sts = [&](int buf) {
        if (Af) {
#pragma unroll
            for (int i = 0; i < 4; ++i) {
                int f = tid + i * 256;
                int r = f >> 3, c4 = (f & 7) << 2;
                bf16* d = &sA[buf][r * LSTR + c4];
                d[0] = cvt_bf16(raf[i].x); d[1] = cvt_bf16(raf[i].y);
                d[2] = cvt_bf16(raf[i].z); d[3] = cvt_bf16(raf[i].w);
            }
        } else {
#pragma unroll
            for (int i = 0; i < 4; ++i) {
                int f = tid + i * 256;
                int r = f >> 3, c4 = (f & 7) << 2;
                *(unsigned long long*)(&sA[buf][r * LSTR + c4]) = rab[i];
            }
        }
        const bf16* rbp = (const bf16*)rbv;
#pragma unroll
        for (int i = 0; i < 32; ++i) sB[buf][(bn0 + i) * LSTR + bk] = rbp[i];
    };

    v8f acc[4][4] = {};

    ldg(0);
    sts(0);
    __syncthreads();
    int cur = 0;

    for (int k0 = 0; k0 < K; k0 += GBK) {
        const bool has_next = (k0 + GBK) < K;
        if (has_next) ldg(k0 + GBK);          // global loads in flight under WMMA

        v16bf af[4];
#pragma unroll
        for (int mi = 0; mi < 4; ++mi)
            af[mi] = frag_ld(sA[cur], LSTR, wm * 64 + mi * 16, 0, lane);
#pragma unroll
        for (int nj = 0; nj < 4; ++nj) {
            v16bf b = frag_ld(sB[cur], LSTR, wn * 64 + nj * 16, 0, lane);
#pragma unroll
            for (int mi = 0; mi < 4; ++mi)
                acc[mi][nj] = wmma_bf16(af[mi], b, acc[mi][nj]);
        }

        if (has_next) {
            sts(cur ^ 1);                     // disjoint buffer: no pre-barrier
            __syncthreads();
            cur ^= 1;
        }
    }

    // ---- epilogue ----
    const int hi = (lane & 16) ? 8 : 0;
    const int nc = lane & 15;
#pragma unroll
    for (int mi = 0; mi < 4; ++mi)
#pragma unroll
        for (int nj = 0; nj < 4; ++nj)
#pragma unroll
            for (int r = 0; r < 8; ++r) {
                size_t row = gm0 + wm * 64 + mi * 16 + r + hi;
                size_t col = gn0 + wn * 64 + nj * 16 + nc;
                float x = acc[mi][nj][r];
                if (gelu) x = 0.5f * x * (1.0f + erff(x * 0.70710678118f));
                if (Cf) Cf[row * N + col] = x;
                else    Cb[row * N + col] = cvt_bf16(x);
            }
}

// ---------------------------------------------------------------------------
// Fused windowed attention. One workgroup (9 waves, 288 threads) per window.
// q,k,v,out: bf16 [4, L, C].  Shift handled by gather/scatter roll; mask
// computed analytically from region ids.
// LDS (dynamic): sQ[144*40] sK[144*40] sP[144*168] sV[256*168] + rid/tokoff
// ---------------------------------------------------------------------------
#define PSTR 168
#define KPAD 160

__global__ __launch_bounds__(288) void
k_attn(const bf16* __restrict__ Q, const bf16* __restrict__ Kb,
       const bf16* __restrict__ V, bf16* __restrict__ O, int is_shift) {
    extern __shared__ char smem[];
    bf16* sQ = (bf16*)smem;                  // 144*40
    bf16* sK = sQ + SWIN * LSTR;             // 144*40
    bf16* sP = sK + SWIN * LSTR;             // 144*168 (probs, K-padded to 160)
    bf16* sV = sP + SWIN * PSTR;             // 256*168 (V transposed: [c][t])
    int* rid    = (int*)(sV + CDIM * PSTR);  // 144
    int* tokoff = rid + SWIN;                // 144

    const int tid  = threadIdx.x;
    const int lane = tid & 31;
    const int w    = tid >> 5;               // wave id 0..8 -> 16 query rows
    const int win  = blockIdx.x;
    const int bi   = win >> 8;
    const int wy   = (win >> 4) & 15;
    const int wx   = win & 15;
    const int sh   = is_shift ? SHFT : 0;

    // ---- phase 0: token offsets, mask region ids, zero K-padding ----
    if (tid < SWIN) {
        int t  = tid;
        int sy = t / WSZ, sx = t % WSZ;
        int iy = (wy * WSZ + sy + sh) % IMG_H;
        int ix = (wx * WSZ + sx + sh) % IMG_W;
        tokoff[t] = ((bi * LTOK) + iy * IMG_W + ix) << 8;   // *CDIM
        int gy = wy * WSZ + sy, gx = wx * WSZ + sx;         // unshifted grid
        int ry = gy < (IMG_H - WSZ) ? 0 : (gy < (IMG_H - SHFT) ? 1 : 2);
        int rx = gx < (IMG_W - WSZ) ? 0 : (gx < (IMG_W - SHFT) ? 1 : 2);
        rid[t] = ry * 3 + rx;
    }
    for (int i = tid; i < SWIN * (KPAD - SWIN); i += 288) {  // sP pad t=144..159
        int s = i >> 4;
        sP[s * PSTR + SWIN + (i & 15)] = (bf16)0.0f;
    }
    for (int i = tid; i < CDIM * (KPAD - SWIN); i += 288) {  // sV pad t=144..159
        int c = i >> 4;
        sV[c * PSTR + SWIN + (i & 15)] = (bf16)0.0f;
    }

    // ---- phase 1: scores = Q K^T / 16 over c-chunks of 32 ----
    v8f sc[9] = {};
    for (int kc = 0; kc < CDIM / 32; ++kc) {
        __syncthreads();
        {
            int t = tid >> 1, half = tid & 1;
            const bf16* qs = Q  + tokoff[t] + kc * 32 + half * 16;
            const bf16* ks = Kb + tokoff[t] + kc * 32 + half * 16;
            bf16* qd = &sQ[t * LSTR + half * 16];
            bf16* kd = &sK[t * LSTR + half * 16];
#pragma unroll
            for (int i = 0; i < 16; ++i) { qd[i] = qs[i]; kd[i] = ks[i]; }
        }
        __syncthreads();
        v16bf a = frag_ld(sQ, LSTR, w * 16, 0, lane);
#pragma unroll
        for (int j = 0; j < 9; ++j) {
            v16bf b = frag_ld(sK, LSTR, j * 16, 0, lane);
            sc[j] = wmma_bf16(a, b, sc[j]);
        }
    }
    __syncthreads();

    // ---- softmax (16 rows per wave, row split across 16 lanes) ----
    const int hi = (lane & 16) ? 8 : 0;
    const int nc = lane & 15;
#pragma unroll
    for (int r = 0; r < 8; ++r) {
        int srow = w * 16 + r + hi;
        int rs = rid[srow];
        float mx = -3.0e38f;
#pragma unroll
        for (int j = 0; j < 9; ++j) {
            float x = sc[j][r] * 0.0625f;                      // 1/sqrt(256)
            if (is_shift && rs != rid[j * 16 + nc]) x -= 100.0f;
            sc[j][r] = x;
            mx = fmaxf(mx, x);
        }
#pragma unroll
        for (int d = 1; d < 16; d <<= 1) mx = fmaxf(mx, __shfl_xor(mx, d, 32));
        float sum = 0.0f;
#pragma unroll
        for (int j = 0; j < 9; ++j) {
            float e = __expf(sc[j][r] - mx);
            sc[j][r] = e;
            sum += e;
        }
#pragma unroll
        for (int d = 1; d < 16; d <<= 1) sum += __shfl_xor(sum, d, 32);
        float inv = 1.0f / sum;
#pragma unroll
        for (int j = 0; j < 9; ++j)
            sP[srow * PSTR + j * 16 + nc] = cvt_bf16(sc[j][r] * inv);
    }
    __syncthreads();

    // ---- phase 2: stage V transposed, then O = P @ V ----
    {
        int t = tid >> 1, half = tid & 1;
        const bf16* vs = V + tokoff[t] + half * 128;
#pragma unroll 8
        for (int i = 0; i < 128; ++i)
            sV[(half * 128 + i) * PSTR + t] = vs[i];
    }
    __syncthreads();

    v16bf pa[5];
#pragma unroll
    for (int kt = 0; kt < 5; ++kt)
        pa[kt] = frag_ld(sP, PSTR, w * 16, kt * 32, lane);

#pragma unroll
    for (int cf = 0; cf < 16; ++cf) {
        v8f o = {};
#pragma unroll
        for (int kt = 0; kt < 5; ++kt) {
            v16bf b = frag_ld(sV, PSTR, cf * 16, kt * 32, lane);
            o = wmma_bf16(pa[kt], b, o);
        }
#pragma unroll
        for (int r = 0; r < 8; ++r) {
            int srow = w * 16 + r + hi;
            O[tokoff[srow] + cf * 16 + nc] = cvt_bf16(o[r]);
        }
    }
}

// ---------------------------------------------------------------------------
// LayerNorm (+residual or concat-builder). One wave per row of 256.
//  mode 0: outf[row] = feat0[row] + LN(proj[row])*g+b
//  mode 1: outx[row][0:256] = bf16(feat0), outx[row][256:512] = bf16(LN(proj))
// ---------------------------------------------------------------------------
__global__ __launch_bounds__(256) void
k_ln_res(const float* __restrict__ proj, const float* __restrict__ g,
         const float* __restrict__ b, const float* __restrict__ feat0,
         float* __restrict__ outf, bf16* __restrict__ outx, int mode, int rows) {
    int row  = blockIdx.x * 8 + (threadIdx.x >> 5);
    int lane = threadIdx.x & 31;
    if (row >= rows) return;
    const float* p = proj + (size_t)row * CDIM;
    float v[8];
    float s = 0.0f;
#pragma unroll
    for (int i = 0; i < 8; ++i) { v[i] = p[lane + i * 32]; s += v[i]; }
#pragma unroll
    for (int d = 1; d < 32; d <<= 1) s += __shfl_xor(s, d, 32);
    float mu = s * (1.0f / CDIM);
    float q = 0.0f;
#pragma unroll
    for (int i = 0; i < 8; ++i) { float d0 = v[i] - mu; q += d0 * d0; }
#pragma unroll
    for (int d = 1; d < 32; d <<= 1) q += __shfl_xor(q, d, 32);
    float rstd = rsqrtf(q * (1.0f / CDIM) + 1e-5f);
#pragma unroll
    for (int i = 0; i < 8; ++i) {
        int c = lane + i * 32;
        float y  = (v[i] - mu) * rstd * g[c] + b[c];
        float f0 = feat0[(size_t)row * CDIM + c];
        if (mode == 0) outf[(size_t)row * CDIM + c] = f0 + y;
        else {
            outx[(size_t)row * 512 + c]        = cvt_bf16(f0);
            outx[(size_t)row * 512 + CDIM + c] = cvt_bf16(y);
        }
    }
}

// ---------------------------------------------------------------------------
// Host orchestration
// ---------------------------------------------------------------------------
extern "C" void kernel_launch(void* const* d_in, const int* in_sizes, int n_in,
                              void* d_out, int out_size, void* d_ws, size_t ws_size,
                              hipStream_t stream) {
    (void)in_sizes; (void)n_in; (void)out_size; (void)ws_size;

    const float* feat_r = (const float*)d_in[0];
    const float* feat_t = (const float*)d_in[1];

    struct Blk { const float *wq,*wk,*wv,*wfc,*ng,*nb,*w1,*w2,*fng,*fnb; };
    Blk blk[4];
    int pi = 2;
    for (int i = 0; i < 4; ++i) {
        blk[i].wq  = (const float*)d_in[pi++];
        blk[i].wk  = (const float*)d_in[pi++];
        blk[i].wv  = (const float*)d_in[pi++];
        blk[i].wfc = (const float*)d_in[pi++];
        blk[i].ng  = (const float*)d_in[pi++];
        blk[i].nb  = (const float*)d_in[pi++];
        if (i & 1) {
            blk[i].w1  = (const float*)d_in[pi++];
            blk[i].w2  = (const float*)d_in[pi++];
            blk[i].fng = (const float*)d_in[pi++];
            blk[i].fnb = (const float*)d_in[pi++];
        } else { blk[i].w1 = blk[i].w2 = blk[i].fng = blk[i].fnb = nullptr; }
    }

    // workspace carve
    char* wp = (char*)d_ws;
    auto bump = [&](size_t bytes) {
        char* r = wp;
        wp += (bytes + 255) & ~(size_t)255;
        return (void*)r;
    };
    float* featA = (float*)bump((size_t)MROW * CDIM * 4);
    float* featB = (float*)bump((size_t)MROW * CDIM * 4);
    bf16*  qb    = (bf16*) bump((size_t)MROW * CDIM * 2);
    bf16*  kb    = (bf16*) bump((size_t)MROW * CDIM * 2);
    bf16*  vb    = (bf16*) bump((size_t)MROW * CDIM * 2);
    bf16*  ab    = (bf16*) bump((size_t)MROW * CDIM * 2);
    float* proj  = (float*)bump((size_t)MROW * CDIM * 4);
    bf16*  xcat  = (bf16*) bump((size_t)MROW * 512 * 2);
    bf16*  hid   = (bf16*) bump((size_t)MROW * 2048 * 2);
    bf16*  wbq   = (bf16*) bump(65536 * 2);
    bf16*  wbk   = (bf16*) bump(65536 * 2);
    bf16*  wbv   = (bf16*) bump(65536 * 2);
    bf16*  wbfc  = (bf16*) bump(65536 * 2);
    bf16*  wb1   = (bf16*) bump((size_t)512 * 2048 * 2);
    bf16*  wb2   = (bf16*) bump((size_t)2048 * CDIM * 2);

    auto cdiv = [](size_t a, size_t b) { return (unsigned)((a + b - 1) / b); };
    auto conv = [&](const float* s, bf16* d, int n) {
        k_f2b<<<cdiv(n, 256), 256, 0, stream>>>(s, d, n);
    };
    auto gemm = [&](const float* Af, const bf16* Ab, const bf16* Bw,
                    float* Cf, bf16* Cb, int M, int N, int K, int gelu) {
        dim3 g(N / GBN, M / GBM);
        k_gemm<<<g, 256, 0, stream>>>(Af, Ab, Bw, Cf, Cb, M, N, K, gelu);
    };

    const size_t FEATN = (size_t)MROW * CDIM;
    k_pack<<<cdiv(FEATN, 256), 256, 0, stream>>>(feat_r, feat_t, featA, featB);

    const size_t attn_lds = (size_t)(SWIN * LSTR + SWIN * LSTR + SWIN * PSTR +
                                     CDIM * PSTR) * 2 + 2 * SWIN * 4;

    for (int b = 0; b < 4; ++b) {
        const bool ffn = (b & 1);
        const int  is_shift = (b >> 1) & 1;

        conv(blk[b].wq, wbq, 65536);
        conv(blk[b].wk, wbk, 65536);
        conv(blk[b].wv, wbv, 65536);
        conv(blk[b].wfc, wbfc, 65536);

        const float* feat1 = ffn ? featB : featA;   // cross vs self K/V source
        gemm(featA, nullptr, wbq, nullptr, qb, MROW, CDIM, CDIM, 0);
        gemm(feat1, nullptr, wbk, nullptr, kb, MROW, CDIM, CDIM, 0);
        gemm(feat1, nullptr, wbv, nullptr, vb, MROW, CDIM, CDIM, 0);

        k_attn<<<B2 * NWIN * NWIN, 288, attn_lds, stream>>>(qb, kb, vb, ab, is_shift);

        gemm(nullptr, ab, wbfc, proj, nullptr, MROW, CDIM, CDIM, 0);

        if (!ffn) {
            k_ln_res<<<MROW / 8, 256, 0, stream>>>(proj, blk[b].ng, blk[b].nb,
                                                   featA, featA, nullptr, 0, MROW);
        } else {
            k_ln_res<<<MROW / 8, 256, 0, stream>>>(proj, blk[b].ng, blk[b].nb,
                                                   featA, nullptr, xcat, 1, MROW);
            conv(blk[b].w1, wb1, 512 * 2048);
            conv(blk[b].w2, wb2, 2048 * CDIM);
            gemm(nullptr, xcat, wb1, nullptr, hid, MROW, 2048, 512, 1);   // +GELU
            gemm(nullptr, hid,  wb2, proj, nullptr, MROW, CDIM, 2048, 0);
            k_ln_res<<<MROW / 8, 256, 0, stream>>>(proj, blk[b].fng, blk[b].fnb,
                                                   featA, featA, nullptr, 0, MROW);
            if (b == 1)
                k_swap<<<cdiv(FEATN, 256), 256, 0, stream>>>(featA, featB);
        }
    }

    k_unpack<<<cdiv(2 * FEATN, 256), 256, 0, stream>>>(featA, (float*)d_out);
}
